// PointGNN_62938450756151
// MI455X (gfx1250) — compile-verified, matrix operand order checked by reading, not run on hardware
//
#include <hip/hip_runtime.h>
#include <cstdint>

// ---- problem constants (from reference) ----
#define B_   4
#define N_   384
#define D_   8
#define T_   3
#define R_   0.05f
#define LN_EPS 1e-5f

// per-layer f16 weight block in workspace: W1t(64x32) | W2t(128x64) | W3t(128x128)
#define WPT  (2048 + 8192 + 16384)   // 26624 halves per t

typedef __attribute__((ext_vector_type(16))) _Float16 v16h;
typedef __attribute__((ext_vector_type(8)))  float    v8f;

// dynamic LDS layout (bytes):
//   [0, 53248)        f16 weights W1t|W2t|W3t
//   then f32: b1(64) g1(64) be1(64) b2(128) g2(128) be2(128) b3(128)
//             pool(128) hb(128) hb2(128) misc(32)
//   then int adj[4*16]
//   then f32 C-scratch 4 waves x 16x128
//   then f16 A-scratch 4 waves x 16x128
#define SMEM_BYTES 107136

struct IterParams {
    // f-MLP (biases + LN params in f32, weights pre-converted f16 in ws)
    const float *f_b1, *f_g1, *f_be1, *f_b2, *f_g2, *f_be2, *f_b3;
    const _Float16 *fW;
    // h-MLP (all f32, used cooperatively)
    const float *h_w1, *h_b1, *h_g1, *h_be1, *h_w2, *h_b2, *h_g2, *h_be2, *h_w3, *h_b3;
    // g-MLP
    const float *g_w1, *g_b1, *g_g1, *g_be1, *g_w2, *g_b2, *g_g2, *g_be2, *g_w3, *g_b3;
};

struct PrepArgs {
    const float *w1[T_], *w2[T_], *w3[T_];
    _Float16 *out;
};

// intra-wave LDS ordering: CDNA5 split counter for DS ops + compiler barrier
#define WAVE_SYNC() asm volatile("s_wait_dscnt 0" ::: "memory")

__device__ __forceinline__ v8f wmma32(v16h a, v16h b, v8f c) {
    return __builtin_amdgcn_wmma_f32_16x16x32_f16(false, a, false, b, (short)0, c, false, false);
}

// Load a 16x32 fp16 fragment (A layout, == B layout for N-major weights) from a
// row-major [rows][stride] array. lane&15 selects row, lane>>4 selects K half.
__device__ __forceinline__ v16h ldfrag(const _Float16* base, int stride, int lane) {
    int r = lane & 15, hh = lane >> 4;
    const _Float16* p = base + r * stride + hh * 8;
    v16h f;
#pragma unroll
    for (int v = 0; v < 8; ++v) {
        int k = ((v >> 2) * 16) + ((v & 3) * 2);
        f[2 * v]     = p[k];
        f[2 * v + 1] = p[k + 1];
    }
    return f;
}

// cooperative LayerNorm + relu over n features (block-wide, thread 0 does stats)
__device__ void block_ln_relu(const float* in, float* out, const float* g,
                              const float* be, int n, int tid, float* mv) {
    __syncthreads();
    if (tid == 0) {
        float m = 0.f;
        for (int k = 0; k < n; ++k) m += in[k];
        m /= (float)n;
        float v = 0.f;
        for (int k = 0; k < n; ++k) { float d = in[k] - m; v += d * d; }
        v /= (float)n;
        mv[0] = m;
        mv[1] = rsqrtf(v + LN_EPS);
    }
    __syncthreads();
    if (tid < n) {
        float y = (in[tid] - mv[0]) * mv[1] * g[tid] + be[tid];
        out[tid] = fmaxf(y, 0.f);
    }
    __syncthreads();
}

// weight prep: f-MLP weights -> f16, transposed (N x Kpad), one block per t
__global__ void pgnn_prep(PrepArgs pa) {
    int t = blockIdx.x;
    const float *w1 = pa.w1[t], *w2 = pa.w2[t], *w3 = pa.w3[t];
    _Float16* o = pa.out + t * WPT;
    for (int e = threadIdx.x; e < 2048; e += blockDim.x) {      // 64 x 32 (K=11 padded)
        int n = e >> 5, k = e & 31;
        o[e] = (k < 11) ? (_Float16)w1[k * 64 + n] : (_Float16)0.f;
    }
    for (int e = threadIdx.x; e < 8192; e += blockDim.x) {      // 128 x 64
        int n = e >> 6, k = e & 63;
        o[2048 + e] = (_Float16)w2[k * 128 + n];
    }
    for (int e = threadIdx.x; e < 16384; e += blockDim.x) {     // 128 x 128
        int n = e >> 7, k = e & 127;
        o[10240 + e] = (_Float16)w3[k * 128 + n];
    }
}

__global__ __launch_bounds__(128) void pgnn_pair(
        const float* __restrict__ orig,   // original state (positions x = orig[...,:3])
        const float* __restrict__ cur,    // current state
        float* __restrict__ outp,         // next state
        const long long* __restrict__ fszp,
        IterParams ip) {
    extern __shared__ char smem[];
    _Float16* sW1 = (_Float16*)smem;          // 2048
    _Float16* sW2 = sW1 + 2048;               // 8192
    _Float16* sW3 = sW2 + 8192;               // 16384
    float* fp = (float*)(smem + 53248);
    float* sB1  = fp; fp += 64;
    float* sG1  = fp; fp += 64;
    float* sBe1 = fp; fp += 64;
    float* sB2  = fp; fp += 128;
    float* sG2  = fp; fp += 128;
    float* sBe2 = fp; fp += 128;
    float* sB3  = fp; fp += 128;
    float* sPool = fp; fp += 128;
    float* sHb   = fp; fp += 128;
    float* sHb2  = fp; fp += 128;
    float* sMisc = fp; fp += 32;              // [0..2]=delta [4..11]=state_i [12..13]=mv
    int*   sAdj  = (int*)fp; fp += 64;        // 4 waves x 16 rows
    float* sC    = fp; fp += 4 * 16 * 128;    // per-wave f32 scratch
    _Float16* sA = (_Float16*)fp;             // per-wave f16 A scratch (4 x 16x128)

    float* sDelta = sMisc;
    float* sSt    = sMisc + 4;
    float* sMv    = sMisc + 12;

    const int tid  = threadIdx.x;
    const int lane = tid & 31;
    const int wave = tid >> 5;
    const int b  = blockIdx.x / N_;
    const int ii = blockIdx.x % N_;
    const int bi0 = b * N_;
    const size_t bi = (size_t)(bi0 + ii);
    const int fsz = (int)fszp[b];

    // ---- stage f-weights + params into LDS, init pool, load state_i ----
    for (int e = tid; e < WPT; e += 128) sW1[e] = ip.fW[e];
    if (tid < 64) { sB1[tid] = ip.f_b1[tid]; sG1[tid] = ip.f_g1[tid]; sBe1[tid] = ip.f_be1[tid]; }
    sB2[tid] = ip.f_b2[tid]; sG2[tid] = ip.f_g2[tid]; sBe2[tid] = ip.f_be2[tid];
    sB3[tid] = ip.f_b3[tid];
    sPool[tid] = 0.f;
    if (tid < 8) sSt[tid] = cur[bi * D_ + tid];
    __syncthreads();

    // ---- delta = mlp_h(state_i)  (cooperative f32) ----
    if (tid < 64) {
        float a = ip.h_b1[tid];
        for (int k = 0; k < 8; ++k) a += sSt[k] * ip.h_w1[k * 64 + tid];
        sHb[tid] = a;
    }
    block_ln_relu(sHb, sHb2, ip.h_g1, ip.h_be1, 64, tid, sMv);
    {
        float a = ip.h_b2[tid];
        for (int k = 0; k < 64; ++k) a += sHb2[k] * ip.h_w2[k * 128 + tid];
        sHb[tid] = a;
    }
    block_ln_relu(sHb, sHb2, ip.h_g2, ip.h_be2, 128, tid, sMv);
    if (tid < 3) {
        float a = ip.h_b3[tid];
        for (int k = 0; k < 128; ++k) a += sHb2[k] * ip.h_w3[k * 3 + tid];
        sDelta[tid] = a;
    }
    __syncthreads();

    const float xi0 = orig[bi * D_ + 0];
    const float xi1 = orig[bi * D_ + 1];
    const float xi2 = orig[bi * D_ + 2];

    float* sCw = sC + wave * 2048;
    _Float16* sAw = sA + wave * 2048;
    float pl[4] = {0.f, 0.f, 0.f, 0.f};

    // ---- main loop: each wave handles 6 of the 24 j-tiles of 16 rows ----
    for (int jt = wave; jt < N_ / 16; jt += 4) {
        const int r = lane & 15;
        const int hh = lane >> 4;
        const int j = jt * 16 + r;
        const float* cj = cur  + (size_t)(bi0 + j) * D_;
        const float* oj = orig + (size_t)(bi0 + j) * D_;
        const float xj0 = oj[0], xj1 = oj[1], xj2 = oj[2];

        // features [11] per row into LDS staging, plus adjacency (original x!)
        if (lane < 16) {
            float* ft = sCw + r * 12;
            ft[0] = sDelta[0] - (xj0 - xi0);
            ft[1] = sDelta[1] - (xj1 - xi1);
            ft[2] = sDelta[2] - (xj2 - xi2);
#pragma unroll
            for (int k = 0; k < 8; ++k) ft[3 + k] = cj[k];
            float dx = xj0 - xi0, dy = xj1 - xi1, dz = xj2 - xi2;
            float d2 = dx * dx + dy * dy + dz * dz;
            sAdj[wave * 16 + r] = (d2 < R_ && ii < fsz && j < fsz) ? 1 : 0;
        }
        WAVE_SYNC();

        // build A0 fragment (16x32, K = 11 real + zero pad)
        v16h a0;
        {
            const float* ft = sCw + r * 12;
#pragma unroll
            for (int v = 0; v < 8; ++v) {
                int k0 = ((v >> 2) * 16) + ((v & 3) * 2) + hh * 8;
                a0[2 * v]     = (k0     < 11) ? (_Float16)ft[k0]     : (_Float16)0.f;
                a0[2 * v + 1] = (k0 + 1 < 11) ? (_Float16)ft[k0 + 1] : (_Float16)0.f;
            }
        }
        WAVE_SYNC();

        // ---- layer1: 16x32 @ 32x64 -> 16x64 (4 WMMAs) ----
        v8f c1[4];
#pragma unroll
        for (int nt = 0; nt < 4; ++nt) {
            v8f cz = {};
            c1[nt] = wmma32(a0, ldfrag(sW1 + nt * 16 * 32, 32, lane), cz);
        }
        {
            const int nl = lane & 15;
#pragma unroll
            for (int nt = 0; nt < 4; ++nt) {
                int n = nt * 16 + nl;
#pragma unroll
                for (int v = 0; v < 8; ++v)
                    sCw[(hh * 8 + v) * 64 + n] = c1[nt][v] + sB1[n];
            }
        }
        WAVE_SYNC();

        // LN1 + relu -> f16 A (16x64)
        if (lane < 16) {
            const float* rowp = sCw + lane * 64;
            float m = 0.f;
            for (int n = 0; n < 64; ++n) m += rowp[n];
            m *= (1.f / 64.f);
            float vv = 0.f;
            for (int n = 0; n < 64; ++n) { float d = rowp[n] - m; vv += d * d; }
            vv *= (1.f / 64.f);
            float rs = rsqrtf(vv + LN_EPS);
            _Float16* arow = sAw + lane * 64;
            for (int n = 0; n < 64; ++n) {
                float y = (rowp[n] - m) * rs * sG1[n] + sBe1[n];
                arow[n] = (_Float16)fmaxf(y, 0.f);
            }
        }
        WAVE_SYNC();

        // ---- layer2: 16x64 @ 64x128 -> 16x128 (16 WMMAs) ----
        {
            v16h a20 = ldfrag(sAw, 64, lane);
            v16h a21 = ldfrag(sAw + 32, 64, lane);
            const int nl = lane & 15;
#pragma unroll
            for (int nt = 0; nt < 8; ++nt) {
                v8f c = {};
                c = wmma32(a20, ldfrag(sW2 + nt * 1024, 64, lane), c);
                c = wmma32(a21, ldfrag(sW2 + nt * 1024 + 32, 64, lane), c);
                int n = nt * 16 + nl;
#pragma unroll
                for (int v = 0; v < 8; ++v)
                    sCw[(hh * 8 + v) * 128 + n] = c[v] + sB2[n];
            }
        }
        WAVE_SYNC();

        // LN2 + relu -> f16 A (16x128)
        if (lane < 16) {
            const float* rowp = sCw + lane * 128;
            float m = 0.f;
            for (int n = 0; n < 128; ++n) m += rowp[n];
            m *= (1.f / 128.f);
            float vv = 0.f;
            for (int n = 0; n < 128; ++n) { float d = rowp[n] - m; vv += d * d; }
            vv *= (1.f / 128.f);
            float rs = rsqrtf(vv + LN_EPS);
            _Float16* arow = sAw + lane * 128;
            for (int n = 0; n < 128; ++n) {
                float y = (rowp[n] - m) * rs * sG2[n] + sBe2[n];
                arow[n] = (_Float16)fmaxf(y, 0.f);
            }
        }
        WAVE_SYNC();

        // ---- layer3: 16x128 @ 128x128 -> 16x128 (32 WMMAs), relu ----
        {
            v16h a3[4];
#pragma unroll
            for (int kc = 0; kc < 4; ++kc) a3[kc] = ldfrag(sAw + kc * 32, 128, lane);
            const int nl = lane & 15;
#pragma unroll
            for (int nt = 0; nt < 8; ++nt) {
                v8f c = {};
#pragma unroll
                for (int kc = 0; kc < 4; ++kc)
                    c = wmma32(a3[kc], ldfrag(sW3 + nt * 2048 + kc * 32, 128, lane), c);
                int n = nt * 16 + nl;
#pragma unroll
                for (int v = 0; v < 8; ++v)
                    sCw[(hh * 8 + v) * 128 + n] = fmaxf(c[v] + sB3[n], 0.f);
            }
        }
        WAVE_SYNC();

        // ---- masked max-pool over this tile's 16 rows ----
#pragma unroll
        for (int rr = 0; rr < 16; ++rr) {
            if (sAdj[wave * 16 + rr]) {
#pragma unroll
                for (int q = 0; q < 4; ++q)
                    pl[q] = fmaxf(pl[q], sCw[rr * 128 + lane * 4 + q]);
            }
        }
        WAVE_SYNC();
    }

    // merge wave-local pools (values >= 0, so int-max == float-max; init 0 matches ref)
#pragma unroll
    for (int q = 0; q < 4; ++q)
        atomicMax((int*)(sPool + lane * 4 + q), __float_as_int(pl[q]));
    __syncthreads();

    // ---- new_state_i = relu(mlp_g(pool)) + state_i (cooperative f32) ----
    if (tid < 64) {
        float a = ip.g_b1[tid];
        for (int k = 0; k < 128; ++k) a += sPool[k] * ip.g_w1[k * 64 + tid];
        sHb[tid] = a;
    }
    block_ln_relu(sHb, sHb2, ip.g_g1, ip.g_be1, 64, tid, sMv);
    if (tid < 32) {
        float a = ip.g_b2[tid];
        for (int k = 0; k < 64; ++k) a += sHb2[k] * ip.g_w2[k * 32 + tid];
        sHb[tid] = a;
    }
    block_ln_relu(sHb, sHb2, ip.g_g2, ip.g_be2, 32, tid, sMv);
    if (tid < 8) {
        float a = ip.g_b3[tid];
        for (int k = 0; k < 32; ++k) a += sHb2[k] * ip.g_w3[k * 8 + tid];
        outp[bi * D_ + tid] = fmaxf(a, 0.f) + cur[bi * D_ + tid];
    }
}

extern "C" void kernel_launch(void* const* d_in, const int* in_sizes, int n_in,
                              void* d_out, int out_size, void* d_ws, size_t ws_size,
                              hipStream_t stream) {
    (void)in_sizes; (void)n_in; (void)out_size; (void)ws_size;
    const float* state0 = (const float*)d_in[0];
    const long long* fszp = (const long long*)d_in[1];   // frame_sz is int64

    _Float16* wbuf = (_Float16*)d_ws;                      // 3*WPT halves (~156KB)
    float* stA = (float*)((char*)d_ws + (160 << 10));
    float* stB = (float*)((char*)d_ws + (224 << 10));

    // params flattened jax-style (dict keys sorted): per t -> f(10), g(10), h(10);
    // per MLP -> l1.b, l1.w, l2.b, l2.w, l3.b, l3.w, n1.be, n1.g, n2.be, n2.g
    PrepArgs pa;
    for (int t = 0; t < T_; ++t) {
        int base = 2 + t * 30;
        pa.w1[t] = (const float*)d_in[base + 1];
        pa.w2[t] = (const float*)d_in[base + 3];
        pa.w3[t] = (const float*)d_in[base + 5];
    }
    pa.out = wbuf;
    pgnn_prep<<<dim3(T_), dim3(256), 0, stream>>>(pa);

    const float* curp = state0;
    for (int t = 0; t < T_; ++t) {
        int base = 2 + t * 30;
        IterParams ip;
        ip.f_b1 = (const float*)d_in[base + 0];
        ip.f_b2 = (const float*)d_in[base + 2];
        ip.f_b3 = (const float*)d_in[base + 4];
        ip.f_be1 = (const float*)d_in[base + 6];
        ip.f_g1  = (const float*)d_in[base + 7];
        ip.f_be2 = (const float*)d_in[base + 8];
        ip.f_g2  = (const float*)d_in[base + 9];
        ip.fW = wbuf + t * WPT;
        ip.g_b1 = (const float*)d_in[base + 10];
        ip.g_w1 = (const float*)d_in[base + 11];
        ip.g_b2 = (const float*)d_in[base + 12];
        ip.g_w2 = (const float*)d_in[base + 13];
        ip.g_b3 = (const float*)d_in[base + 14];
        ip.g_w3 = (const float*)d_in[base + 15];
        ip.g_be1 = (const float*)d_in[base + 16];
        ip.g_g1  = (const float*)d_in[base + 17];
        ip.g_be2 = (const float*)d_in[base + 18];
        ip.g_g2  = (const float*)d_in[base + 19];
        ip.h_b1 = (const float*)d_in[base + 20];
        ip.h_w1 = (const float*)d_in[base + 21];
        ip.h_b2 = (const float*)d_in[base + 22];
        ip.h_w2 = (const float*)d_in[base + 23];
        ip.h_b3 = (const float*)d_in[base + 24];
        ip.h_w3 = (const float*)d_in[base + 25];
        ip.h_be1 = (const float*)d_in[base + 26];
        ip.h_g1  = (const float*)d_in[base + 27];
        ip.h_be2 = (const float*)d_in[base + 28];
        ip.h_g2  = (const float*)d_in[base + 29];

        float* nxt = (t == 0) ? stA : (t == 1) ? stB : (float*)d_out;
        pgnn_pair<<<dim3(B_ * N_), dim3(128), SMEM_BYTES, stream>>>(
            state0, curp, nxt, fszp, ip);
        curp = nxt;
    }
}
